// MoE_3401614099130
// MI455X (gfx1250) — matrix operand order repaired and checked by reading, not compiled, not run on hardware
//
#include <hip/hip_runtime.h>
#include <cstdint>
#include <cstddef>

// ---------------------------------------------------------------------------
// MoE (SwiGLU, top-2 of 8 experts) for gfx1250 using bf16 WMMA.
// D=1024, H=2048, E=8, K=2, N=8192 tokens. y fp32 + aux loss scalar.
// GEMM tiles: 128x128 block, 8 waves as 4Mx2N, wave tile 32x64, K-step 32.
// ---------------------------------------------------------------------------

typedef __bf16 bf16;
typedef __attribute__((ext_vector_type(16))) __bf16 v16bf;
typedef __attribute__((ext_vector_type(4)))  __bf16 v4bf;
typedef __attribute__((ext_vector_type(8)))  float  v8f;

#define D_DIM   1024
#define H_DIM   2048
#define E_NUM   8
#define N_TOK   8192
#define PAD_R   128              // per-expert row padding granularity (= TM)
#define SLOT_CAP 17408           // N*K + E*127, rounded up to 128
#define AUX_COEF 0.01f

// ---- workspace layout (bytes, all 256-aligned) ----------------------------
static constexpr size_t OFF_WB1T  = 0;                       // [E][H][D] bf16
static constexpr size_t OFF_WB3T  = OFF_WB1T + (size_t)E_NUM*H_DIM*D_DIM*2;
static constexpr size_t OFF_WB2T  = OFF_WB3T + (size_t)E_NUM*H_DIM*D_DIM*2; // [E][D][H]
static constexpr size_t OFF_XB    = OFF_WB2T + (size_t)E_NUM*H_DIM*D_DIM*2; // [N][D] bf16
static constexpr size_t OFF_HBUF  = OFF_XB   + (size_t)N_TOK*D_DIM*2;       // [SLOT_CAP][H] bf16
static constexpr size_t OFF_TOKE  = OFF_HBUF + (size_t)SLOT_CAP*H_DIM*2;    // [N][2] int
static constexpr size_t OFF_TOKW  = OFF_TOKE + (size_t)N_TOK*2*4;           // [N][2] f32
static constexpr size_t OFF_STOK  = OFF_TOKW + (size_t)N_TOK*2*4;           // [SLOT_CAP] int
static constexpr size_t OFF_SW    = OFF_STOK + (size_t)SLOT_CAP*4;          // [SLOT_CAP] f32
static constexpr size_t OFF_CNT   = OFF_SW   + (size_t)SLOT_CAP*4;          // [E] u32
static constexpr size_t OFF_FILL  = OFF_CNT  + 256;                         // [E] u32
static constexpr size_t OFF_BASE  = OFF_FILL + 256;                         // [E+1] int
static constexpr size_t OFF_GATEP = OFF_BASE + 256;                         // [1024][8] f32

// ---------------------------------------------------------------------------
__device__ __forceinline__ v16bf load_frag(const bf16* p0, const bf16* p1) {
  union { uint4 u[2]; v16bf v; } r;
  r.u[0] = *(const uint4*)p0;
  r.u[1] = *(const uint4*)p1;
  return r.v;
}

__device__ __forceinline__ v8f wmma_bf16(v16bf a, v16bf b, v8f c) {
  return __builtin_amdgcn_wmma_f32_16x16x32_bf16(false, a, false, b,
                                                 (short)0, c, false, false);
}

// ---- 0. zero y + counters -------------------------------------------------
__global__ void zero_kernel(float4* y4, unsigned* counts, unsigned* fill) {
  size_t i = (size_t)blockIdx.x * 256 + threadIdx.x;
  if (i < (size_t)N_TOK * D_DIM / 4) y4[i] = make_float4(0.f, 0.f, 0.f, 0.f);
  if (i < E_NUM) { counts[i] = 0u; fill[i] = 0u; }
}

// ---- 1. fp32 [E][R][C] -> bf16 [E][C][R] (tiled transpose + convert) ------
__global__ void transpose_cvt(const float* __restrict__ in, bf16* __restrict__ out,
                              int R, int C) {
  __shared__ float t[32][33];
  const int e  = blockIdx.z;
  const float* I = in  + (size_t)e * R * C;
  bf16*        O = out + (size_t)e * R * C;
  const int c0 = blockIdx.x * 32, r0 = blockIdx.y * 32;
  const int tx = threadIdx.x, ty = threadIdx.y;   // (32,8)
  #pragma unroll
  for (int i = 0; i < 32; i += 8)
    t[ty + i][tx] = I[(size_t)(r0 + ty + i) * C + (c0 + tx)];
  __syncthreads();
  #pragma unroll
  for (int i = 0; i < 32; i += 8)
    O[(size_t)(c0 + ty + i) * R + (r0 + tx)] = (bf16)t[tx][ty + i];
}

// ---- 2. x fp32 -> bf16 ----------------------------------------------------
__global__ void cvt_x_kernel(const float* __restrict__ in, bf16* __restrict__ out) {
  size_t i = ((size_t)blockIdx.x * 256 + threadIdx.x) * 4;
  if (i + 3 < (size_t)N_TOK * D_DIM) {
    float4 f = *(const float4*)(in + i);
    v4bf o; o[0] = (bf16)f.x; o[1] = (bf16)f.y; o[2] = (bf16)f.z; o[3] = (bf16)f.w;
    *(v4bf*)(out + i) = o;
  }
}

// ---- 3. gate: softmax + top-2, counts, deterministic score partials -------
__global__ void gate_kernel(const float* __restrict__ x, const float* __restrict__ Wg,
                            int* __restrict__ tokExp, float* __restrict__ tokW,
                            unsigned* __restrict__ counts, float* __restrict__ gateP) {
  __shared__ float sc[8][E_NUM];
  const int wave = threadIdx.x >> 5, lane = threadIdx.x & 31;
  const int tok = blockIdx.x * 8 + wave;
  const float* xp = x + (size_t)tok * D_DIM;
  float acc[E_NUM];
  #pragma unroll
  for (int e = 0; e < E_NUM; e++) acc[e] = 0.f;
  for (int i = lane; i < D_DIM; i += 32) {
    const float xv = xp[i];
    #pragma unroll
    for (int e = 0; e < E_NUM; e++) acc[e] += xv * Wg[e * D_DIM + i];
  }
  #pragma unroll
  for (int e = 0; e < E_NUM; e++)
    for (int m = 16; m > 0; m >>= 1) acc[e] += __shfl_xor(acc[e], m, 32);
  if (lane == 0) {
    float mx = acc[0];
    #pragma unroll
    for (int e = 1; e < E_NUM; e++) mx = fmaxf(mx, acc[e]);
    float p[E_NUM], s = 0.f;
    #pragma unroll
    for (int e = 0; e < E_NUM; e++) { p[e] = __expf(acc[e] - mx); s += p[e]; }
    const float inv = 1.f / s;
    #pragma unroll
    for (int e = 0; e < E_NUM; e++) { p[e] *= inv; sc[wave][e] = p[e]; }
    int i0 = 0;
    #pragma unroll
    for (int e = 1; e < E_NUM; e++) if (p[e] > p[i0]) i0 = e;      // lax.top_k ties: lowest idx
    int i1 = (i0 == 0) ? 1 : 0;
    #pragma unroll
    for (int e = 0; e < E_NUM; e++) if (e != i0 && p[e] > p[i1]) i1 = e;
    tokExp[tok * 2 + 0] = i0; tokExp[tok * 2 + 1] = i1;
    tokW[tok * 2 + 0] = p[i0]; tokW[tok * 2 + 1] = p[i1];
    atomicAdd(&counts[i0], 1u);
    atomicAdd(&counts[i1], 1u);
  }
  __syncthreads();
  if (threadIdx.x < E_NUM) {       // fixed-order per-block reduction -> deterministic
    float s = 0.f;
    #pragma unroll
    for (int t = 0; t < 8; t++) s += sc[t][threadIdx.x];
    gateP[blockIdx.x * E_NUM + threadIdx.x] = s;
  }
}

// ---- 4. padded per-expert offsets -----------------------------------------
__global__ void offsets_kernel(const unsigned* __restrict__ counts, int* __restrict__ eBase) {
  if (threadIdx.x == 0) {
    int off = 0;
    for (int e = 0; e < E_NUM; e++) {
      eBase[e] = off;
      off += (int)((counts[e] + (PAD_R - 1u)) & ~(unsigned)(PAD_R - 1));
    }
    eBase[E_NUM] = off;
  }
}

// ---- 5. dispatch tokens into per-expert slot lists ------------------------
__global__ void dispatch_kernel(const int* __restrict__ tokExp, const float* __restrict__ tokW,
                                const int* __restrict__ eBase, unsigned* __restrict__ fill,
                                int* __restrict__ slotTok, float* __restrict__ slotW) {
  const int n = blockIdx.x * 256 + threadIdx.x;
  if (n >= N_TOK) return;
  #pragma unroll
  for (int k = 0; k < 2; k++) {
    const int e = tokExp[n * 2 + k];
    const unsigned pos = atomicAdd(&fill[e], 1u);
    const int s = eBase[e] + (int)pos;
    slotTok[s] = n;
    slotW[s] = tokW[n * 2 + k];
  }
}

// ---- 6. GEMM1: h = silu(X W1 + b1) * (X W3 + b3), bf16 out ----------------
// 128x128 tile: waves 4Mx2N, wave tile 32x64 -> 2x4 WMMA tiles per matrix.
__global__ __launch_bounds__(256) void gemm1_kernel(
    const bf16* __restrict__ xb, const bf16* __restrict__ w1T, const bf16* __restrict__ w3T,
    const float* __restrict__ b1, const float* __restrict__ b3,
    const int* __restrict__ eBase, const unsigned* __restrict__ counts,
    const int* __restrict__ slotTok, bf16* __restrict__ hbuf) {
  const int e = blockIdx.z;
  const int cnt = (int)counts[e];
  const int padded = (cnt + (PAD_R - 1)) & ~(PAD_R - 1);
  const int rt = blockIdx.y;
  if (rt * PAD_R >= padded) return;              // block-uniform exit
  const int rbase = eBase[e];
  const int ct = blockIdx.x;

  __shared__ bf16 lA[128 * 40];                  // [row][k], stride 40 (pad)
  __shared__ bf16 lB1[128 * 40];                 // [n(h)][k], stride 40
  __shared__ bf16 lB3[128 * 40];

  const int tid = threadIdx.x;
  const int lane = tid & 31, wv = tid >> 5;
  const int waveM = wv >> 1, waveN = wv & 1;     // 4 x 2 waves
  const int laneLo = lane & 15, laneHi = lane >> 4;

  // A stage: 128 rows x 2 chunks of 16 bf16 (32B each)
  const int ar = tid >> 1, ac = (tid & 1) * 16;
  const int slotR = rt * PAD_R + ar;
  const bf16* aSrc = nullptr;
  if (slotR < cnt) aSrc = xb + (size_t)slotTok[rbase + slotR] * D_DIM + ac;

  // B stage: 128 rows x 2 chunks of 16 bf16 per matrix
  const int bn = tid >> 1, bk = (tid & 1) * 16;
  const size_t brow = ((size_t)e * H_DIM + (size_t)ct * 128 + bn) * D_DIM + bk;
  const bf16* b1Src = w1T + brow;
  const bf16* b3Src = w3T + brow;

  v8f acc1[2][4] = {};
  v8f acc3[2][4] = {};

  for (int kt = 0; kt < D_DIM; kt += 32) {
    __syncthreads();
    uint4 av0 = make_uint4(0u, 0u, 0u, 0u), av1 = make_uint4(0u, 0u, 0u, 0u);
    if (aSrc) {
      av0 = *(const uint4*)(aSrc + kt);
      av1 = *(const uint4*)(aSrc + kt + 8);
    }
    *(uint4*)&lA[ar * 40 + ac]     = av0;
    *(uint4*)&lA[ar * 40 + ac + 8] = av1;
    *(uint4*)&lB1[bn * 40 + bk]     = *(const uint4*)(b1Src + kt);
    *(uint4*)&lB1[bn * 40 + bk + 8] = *(const uint4*)(b1Src + kt + 8);
    *(uint4*)&lB3[bn * 40 + bk]     = *(const uint4*)(b3Src + kt);
    *(uint4*)&lB3[bn * 40 + bk + 8] = *(const uint4*)(b3Src + kt + 8);
    __syncthreads();
    #pragma unroll
    for (int mi = 0; mi < 2; mi++) {
      // A frag: lanes 0-15 K{0..15}; lanes 16-31 K{16..31}
      const bf16* ap = &lA[(waveM * 32 + mi * 16 + laneLo) * 40 + laneHi * 16];
      const v16bf a = load_frag(ap, ap + 8);
      #pragma unroll
      for (int ni = 0; ni < 4; ni++) {
        // B frag: per-lane column N, contiguous K {0..15}/{16..31}
        const bf16* bp1 = &lB1[(waveN * 64 + ni * 16 + laneLo) * 40 + laneHi * 16];
        acc1[mi][ni] = wmma_bf16(a, load_frag(bp1, bp1 + 8), acc1[mi][ni]);
        const bf16* bp3 = &lB3[(waveN * 64 + ni * 16 + laneLo) * 40 + laneHi * 16];
        acc3[mi][ni] = wmma_bf16(a, load_frag(bp3, bp3 + 8), acc3[mi][ni]);
      }
    }
  }
  #pragma unroll
  for (int mi = 0; mi < 2; mi++) {
    #pragma unroll
    for (int ni = 0; ni < 4; ni++) {
      const int row0 = rt * PAD_R + waveM * 32 + mi * 16 + 8 * laneHi;
      const int col  = ct * 128 + waveN * 64 + ni * 16 + laneLo;
      const float bb1 = b1[e * H_DIM + col];
      const float bb3 = b3[e * H_DIM + col];
      #pragma unroll
      for (int v = 0; v < 8; v++) {
        const float a1 = acc1[mi][ni][v] + bb1;
        const float a3 = acc3[mi][ni][v] + bb3;
        const float hv = a1 / (1.f + __expf(-a1)) * a3;   // silu(a1)*a3
        hbuf[(size_t)(rbase + row0 + v) * H_DIM + col] = (bf16)hv;
      }
    }
  }
}

// ---- 7. GEMM2: y += cw * (h W2 + b2), scatter by token (2 commutative adds)
__global__ __launch_bounds__(256) void gemm2_kernel(
    const bf16* __restrict__ hbuf, const bf16* __restrict__ w2T,
    const float* __restrict__ b2,
    const int* __restrict__ eBase, const unsigned* __restrict__ counts,
    const int* __restrict__ slotTok, const float* __restrict__ slotW,
    float* __restrict__ y) {
  const int e = blockIdx.z;
  const int cnt = (int)counts[e];
  const int padded = (cnt + (PAD_R - 1)) & ~(PAD_R - 1);
  const int rt = blockIdx.y;
  if (rt * PAD_R >= padded) return;
  const int rbase = eBase[e];
  const int ct = blockIdx.x;

  __shared__ bf16 lA[128 * 40];
  __shared__ bf16 lB[128 * 40];

  const int tid = threadIdx.x;
  const int lane = tid & 31, wv = tid >> 5;
  const int waveM = wv >> 1, waveN = wv & 1;     // 4 x 2 waves
  const int laneLo = lane & 15, laneHi = lane >> 4;

  const int ar = tid >> 1, ac = (tid & 1) * 16;
  const bf16* aSrc = hbuf + (size_t)(rbase + rt * PAD_R + ar) * H_DIM + ac;
  const int bn = tid >> 1, bk = (tid & 1) * 16;
  const bf16* bSrc = w2T + ((size_t)e * D_DIM + (size_t)ct * 128 + bn) * H_DIM + bk;

  v8f acc[2][4] = {};

  for (int kt = 0; kt < H_DIM; kt += 32) {
    __syncthreads();
    *(uint4*)&lA[ar * 40 + ac]     = *(const uint4*)(aSrc + kt);
    *(uint4*)&lA[ar * 40 + ac + 8] = *(const uint4*)(aSrc + kt + 8);
    *(uint4*)&lB[bn * 40 + bk]     = *(const uint4*)(bSrc + kt);
    *(uint4*)&lB[bn * 40 + bk + 8] = *(const uint4*)(bSrc + kt + 8);
    __syncthreads();
    #pragma unroll
    for (int mi = 0; mi < 2; mi++) {
      const bf16* ap = &lA[(waveM * 32 + mi * 16 + laneLo) * 40 + laneHi * 16];
      const v16bf a = load_frag(ap, ap + 8);
      #pragma unroll
      for (int ni = 0; ni < 4; ni++) {
        const bf16* bp = &lB[(waveN * 64 + ni * 16 + laneLo) * 40 + laneHi * 16];
        acc[mi][ni] = wmma_bf16(a, load_frag(bp, bp + 8), acc[mi][ni]);
      }
    }
  }
  #pragma unroll
  for (int mi = 0; mi < 2; mi++) {
    #pragma unroll
    for (int ni = 0; ni < 4; ni++) {
      const int row0 = rt * PAD_R + waveM * 32 + mi * 16 + 8 * laneHi;
      const int col  = ct * 128 + waveN * 64 + ni * 16 + laneLo;
      const float bb2 = b2[e * D_DIM + col];
      #pragma unroll
      for (int v = 0; v < 8; v++) {
        const int r = row0 + v;
        if (r < cnt) {
          const int s = rbase + r;
          const float val = (acc[mi][ni][v] + bb2) * slotW[s];
          atomicAdd(&y[(size_t)slotTok[s] * D_DIM + col], val);
        }
      }
    }
  }
}

// ---- 8. aux loss ----------------------------------------------------------
__global__ void aux_kernel(const float* __restrict__ gateP,
                           const unsigned* __restrict__ counts, float* __restrict__ out) {
  __shared__ float P[E_NUM];
  const int t = threadIdx.x;
  if (t < E_NUM) {
    float s = 0.f;
    for (int b = 0; b < N_TOK / 8; b++) s += gateP[b * E_NUM + t];  // fixed order
    P[t] = s / (float)N_TOK;
  }
  __syncthreads();
  if (t == 0) {
    float aux = 0.f;
    for (int e = 0; e < E_NUM; e++)
      aux += ((float)counts[e] / (float)(N_TOK * 2)) * P[e];
    out[(size_t)N_TOK * D_DIM] = AUX_COEF * (float)E_NUM * aux;
  }
}

// ---------------------------------------------------------------------------
extern "C" void kernel_launch(void* const* d_in, const int* in_sizes, int n_in,
                              void* d_out, int out_size, void* d_ws, size_t ws_size,
                              hipStream_t stream) {
  (void)in_sizes; (void)n_in; (void)out_size; (void)ws_size;
  const float* x  = (const float*)d_in[0];
  const float* Wg = (const float*)d_in[1];
  const float* W1 = (const float*)d_in[2];
  const float* b1 = (const float*)d_in[3];
  const float* W2 = (const float*)d_in[4];
  const float* b2 = (const float*)d_in[5];
  const float* W3 = (const float*)d_in[6];
  const float* b3 = (const float*)d_in[7];
  float* y = (float*)d_out;

  char* ws = (char*)d_ws;
  bf16*     wb1T    = (bf16*)(ws + OFF_WB1T);
  bf16*     wb3T    = (bf16*)(ws + OFF_WB3T);
  bf16*     wb2T    = (bf16*)(ws + OFF_WB2T);
  bf16*     xb      = (bf16*)(ws + OFF_XB);
  bf16*     hbuf    = (bf16*)(ws + OFF_HBUF);
  int*      tokExp  = (int*)(ws + OFF_TOKE);
  float*    tokW    = (float*)(ws + OFF_TOKW);
  int*      slotTok = (int*)(ws + OFF_STOK);
  float*    slotW   = (float*)(ws + OFF_SW);
  unsigned* counts  = (unsigned*)(ws + OFF_CNT);
  unsigned* fill    = (unsigned*)(ws + OFF_FILL);
  int*      eBase   = (int*)(ws + OFF_BASE);
  float*    gateP   = (float*)(ws + OFF_GATEP);

  zero_kernel<<<(N_TOK * D_DIM / 4 + 255) / 256, 256, 0, stream>>>((float4*)y, counts, fill);

  // W1,W3: [E,1024,2048] -> [E,2048,1024] ; W2: [E,2048,1024] -> [E,1024,2048]
  transpose_cvt<<<dim3(H_DIM / 32, D_DIM / 32, E_NUM), dim3(32, 8), 0, stream>>>(W1, wb1T, D_DIM, H_DIM);
  transpose_cvt<<<dim3(H_DIM / 32, D_DIM / 32, E_NUM), dim3(32, 8), 0, stream>>>(W3, wb3T, D_DIM, H_DIM);
  transpose_cvt<<<dim3(D_DIM / 32, H_DIM / 32, E_NUM), dim3(32, 8), 0, stream>>>(W2, wb2T, H_DIM, D_DIM);
  cvt_x_kernel<<<(N_TOK * D_DIM / 4 + 255) / 256, 256, 0, stream>>>(x, xb);

  gate_kernel<<<N_TOK / 8, 256, 0, stream>>>(x, Wg, tokExp, tokW, counts, gateP);
  offsets_kernel<<<1, 32, 0, stream>>>(counts, eBase);
  dispatch_kernel<<<N_TOK / 256, 256, 0, stream>>>(tokExp, tokW, eBase, fill, slotTok, slotW);

  gemm1_kernel<<<dim3(H_DIM / 128, N_TOK / PAD_R, E_NUM), 256, 0, stream>>>(
      xb, wb1T, wb3T, b1, b3, eBase, counts, slotTok, hbuf);
  gemm2_kernel<<<dim3(D_DIM / 128, N_TOK / PAD_R, E_NUM), 256, 0, stream>>>(
      hbuf, wb2T, b2, eBase, counts, slotTok, slotW, y);

  aux_kernel<<<1, 256, 0, stream>>>(gateP, counts, y);
}
